// ProjectWDepth_36318243455249
// MI455X (gfx1250) — matrix operand order repaired and checked by reading, not compiled, not run on hardware
//
#include <hip/hip_runtime.h>
#include <math.h>

// BEV voxel-projection as a one-hot selection GEMM on CDNA5 WMMA.
// out[b,c,z,x] = sum_{p : point p lands in winning voxel of cell (z,x)} features[b,c,p]
// D(128x1024) = F(128x256) x S(256x1024), S is {0,1}. Split-f16 trick keeps ~f32 precision:
// F = Fhi + Flo (each f16), C += Fhi*S ; C += Flo*S  with f32 accumulation.
// S is ~0.1% dense -> skip empty (n-tile, k-step) fragments with *scalar* branches
// (condition is wave-uniform), keeping EXEC all-ones around every WMMA.

typedef __attribute__((ext_vector_type(16))) _Float16 v16h;
typedef __attribute__((ext_vector_type(8)))  float    v8f;

#define NPTS   256          // 16x16 image points per batch
#define NCH    128          // channels
#define MAPSZ  32
#define NCELL  (MAPSZ*MAPSZ)   // 1024
#define HMAXV  14
#define CAMH   1.72f
#define CHUNK  256          // cells handled per workgroup
#define NTILES (CHUNK/16)   // 16 N-tiles of 16 cells

__global__ __launch_bounds__(256) void bev_voxel_wmma(
    const float* __restrict__ features,   // (64,128,16,16)
    const float* __restrict__ depth,      // (64,1,16,16)
    const float* __restrict__ cam,        // (3,256)
    float* __restrict__ out)              // (64,128,32,32)
{
    __shared__ int      s_best[CHUNK];    // winning y per cell in this chunk, -1 = empty
    __shared__ int      s_sel[NPTS];      // per point: local winning cell index or -1
    __shared__ unsigned s_tmask[NTILES];  // bit s set iff fragment (tt,s) has any nonzero

    const int blk     = blockIdx.x;
    const int b       = blk >> 2;        // batch
    const int chunk   = blk & 3;         // which 256-cell chunk of the 1024 cells
    const int cell_lo = chunk * CHUNK;

    const int t = threadIdx.x;           // 256 threads = 8 wave32
    const float CELLSZ = 3.2f / 32.0f;   // == f32(0.1), matches reference

    // ---------------- Phase 1: voxelize 256 points, pick winning y per cell ----
    if (t < CHUNK) s_best[t] = -1;
    if (t < NTILES) s_tmask[t] = 0u;
    __syncthreads();

    int my_cell = -1, my_y = -1;
    {
        const int p = t;                                  // one point per thread
        const float d  = depth[b * NPTS + p];
        const float px = d * cam[p];
        const float py = d * cam[NPTS + p] + CAMH;
        const float pz = d * cam[2 * NPTS + p];
        const int xi = (int)floorf(px / CELLSZ) + MAPSZ / 2;
        const int yi = (int)floorf(py / CELLSZ);
        const int zi = (int)floorf(pz / CELLSZ) + MAPSZ;
        const bool valid = (xi >= 0) & (xi < MAPSZ) & (zi >= 0) & (zi < MAPSZ)
                         & (yi >= 0) & (yi < HMAXV);      // data regime: yi >= 1 always
        const int cell = zi * MAPSZ + xi;
        if (valid && cell >= cell_lo && cell < cell_lo + CHUNK) {
            my_cell = cell - cell_lo;
            my_y    = yi;
            atomicMax(&s_best[my_cell], yi);
        }
    }
    __syncthreads();
    int sel = -1;
    if (my_cell >= 0 && s_best[my_cell] == my_y) sel = my_cell;  // point is a winner
    s_sel[t] = sel;
    if (sel >= 0) atomicOr(&s_tmask[sel >> 4], 1u << (t >> 5));  // mark (tt = sel/16, s = p/32)
    __syncthreads();

    // ---------------- Phase 2: split-f16 WMMA GEMM  D = F * S --------------------
    const int wave = t >> 5;         // 8 waves -> 8 M-tiles (16 channels each)
    const int lane = t & 31;
    const int m    = lane & 15;      // A-matrix row within tile (both lane halves cover M=0..15)
    const int hi   = lane >> 4;      // lane-half selects K sub-range
    const int c0   = wave * 16;

    // A fragments for all K=256, per documented 16-bit A 16x32 layout:
    //   element j<8  -> K = k0 + j + 8*hi
    //   element j>=8 -> K = k0 + 16 + (j-8) + 8*hi
    v16h ahi[8], alo[8];
    const float* frow = features + (size_t)(b * NCH + c0 + m) * NPTS;
    #pragma unroll
    for (int s = 0; s < 8; ++s) {
        const int k0 = 32 * s;
        const float* p1 = frow + k0 + 8 * hi;
        const float* p2 = frow + k0 + 16 + 8 * hi;
        #pragma unroll
        for (int j = 0; j < 8; ++j) {
            const float f1 = p1[j];
            const _Float16 h1 = (_Float16)f1;
            ahi[s][j]     = h1;
            alo[s][j]     = (_Float16)(f1 - (float)h1);
            const float f2 = p2[j];
            const _Float16 h2 = (_Float16)f2;
            ahi[s][j + 8] = h2;
            alo[s][j + 8] = (_Float16)(f2 - (float)h2);
        }
    }

    const _Float16 ONE  = (_Float16)1.0f;
    const _Float16 ZERO = (_Float16)0.0f;
    const int n = lane & 15;         // B/C column within tile

    for (int tt = 0; tt < NTILES; ++tt) {
        const int cellg = cell_lo + tt * 16 + n;
        float* outcol = out + (size_t)b * NCH * NCELL + cellg;

        // Wave-uniform fragment-occupancy mask for this n-tile (scalar branch only:
        // EXEC stays all-ones, as WMMA requires).
        unsigned tmask = __builtin_amdgcn_readfirstlane(s_tmask[tt]);

        if (tmask == 0u) {
            // Entire 16x16 output tile is zero: skip all 16 WMMAs.
            #pragma unroll
            for (int r = 0; r < 8; ++r) {
                const int c = c0 + r + 8 * hi;
                __builtin_nontemporal_store(0.0f, outcol + (size_t)c * NCELL);
            }
            continue;
        }

        const int ncol = tt * 16 + n;        // local cell this lane's column maps to
        v8f acc = {};
        #pragma unroll
        for (int s = 0; s < 8; ++s) {
            if ((tmask >> s) & 1u) {         // scalar: skip all-zero 32x16 S fragments
                const int k0 = 32 * s;
                // B fragment (32x16 one-hot), element j<8 -> K = j + 8*hi, j>=8 -> K = 16 + (j-8) + 8*hi
                v16h bmat;
                #pragma unroll
                for (int j = 0; j < 16; ++j) {
                    const int pidx = k0 + 8 * hi + (j < 8 ? j : j + 8);
                    bmat[j] = (s_sel[pidx] == ncol) ? ONE : ZERO;
                }
                acc = __builtin_amdgcn_wmma_f32_16x16x32_f16(
                          false, ahi[s], false, bmat, (short)0, acc, false, false);
                acc = __builtin_amdgcn_wmma_f32_16x16x32_f16(
                          false, alo[s], false, bmat, (short)0, acc, false, false);
            }
        }
        // C/D layout: VGPR r, lane-half hi -> channel row c0 + r + 8*hi, column n
        #pragma unroll
        for (int r = 0; r < 8; ++r) {
            const int c = c0 + r + 8 * hi;
            __builtin_nontemporal_store(acc[r], outcol + (size_t)c * NCELL);
        }
    }
}

extern "C" void kernel_launch(void* const* d_in, const int* in_sizes, int n_in,
                              void* d_out, int out_size, void* d_ws, size_t ws_size,
                              hipStream_t stream) {
    const float* features = (const float*)d_in[0];   // (64,128,16,16) f32
    const float* depth    = (const float*)d_in[1];   // (64,1,16,16)  f32
    const float* cam      = (const float*)d_in[2];   // (3,256)       f32
    float* out            = (float*)d_out;           // (64,128,32,32) f32

    dim3 grid(64 * 4);   // batch x cell-chunk
    dim3 block(256);     // 8 wave32
    hipLaunchKernelGGL(bev_voxel_wmma, grid, block, 0, stream,
                       features, depth, cam, out);
}
